// PCFG_57002805953312
// MI455X (gfx1250) — matrix operand order repaired and checked by reading, not compiled, not run on hardware
//
#include <hip/hip_runtime.h>

typedef __attribute__((ext_vector_type(16))) _Float16 v16h;
typedef __attribute__((ext_vector_type(8)))  _Float16 h8;
typedef __attribute__((ext_vector_type(8)))  float    v8f;

#define NTC   30
#define TTC   60
#define SSC   90
#define NNC   32
#define BBC   16
#define KP    8640            // 90 x-blocks * 96 padded c ; 270 k-steps of 32
#define KSTEPS (KP / 32)      // 270
#define ELSTR 96              // eL LDS row stride (halfs)
#define ERSTR 104             // eR LDS row stride (halfs): 16B aligned, bank-clean
#define NEGV  (-1000000000.0f)

__device__ __forceinline__ float wave_max(float v) {
#pragma unroll
  for (int off = 16; off > 0; off >>= 1)
    v = fmaxf(v, __shfl_xor(v, off, 32));
  return v;
}
__device__ __forceinline__ float wave_sum(float v) {
#pragma unroll
  for (int off = 16; off > 0; off >>= 1)
    v += __shfl_xor(v, off, 32);
  return v;
}

// ---------------------------------------------------------------------------
// beta[b,i,j,c] init: NEG everywhere, diagonal terminals from unary
// ---------------------------------------------------------------------------
__global__ void k_init_beta(const float* __restrict__ unary,
                            float* __restrict__ beta) {
  int idx = blockIdx.x * blockDim.x + threadIdx.x;
  const int total = BBC * NNC * NNC * SSC;
  if (idx >= total) return;
  int c = idx % SSC;
  int j = (idx / SSC) % NNC;
  int i = (idx / (SSC * NNC)) % NNC;
  int b = idx / (SSC * NNC * NNC);
  float v = NEGV;
  if (i == j && c >= NTC) v = unary[(b * NNC + i) * TTC + (c - NTC)];
  beta[idx] = v;
}

// ---------------------------------------------------------------------------
// Per (b,a): mRule_a = max rule ; eruleH[b][a][x*96+c] = fp16 exp(rule - max)
// c in [90,96) zero pad; rows a in [30,32) zero pad.
// ---------------------------------------------------------------------------
__global__ void k_prep_rule(const float* __restrict__ rule,
                            _Float16* __restrict__ eruleH,
                            float* __restrict__ mruleA) {
  int a = blockIdx.x;           // 0..31
  int b = blockIdx.y;
  int tid = threadIdx.x;        // 256 threads
  long hbase = ((long)(b * 32 + a)) * KP;
  if (a >= NTC) {
    for (int k = tid; k < KP; k += 256) eruleH[hbase + k] = (_Float16)0.0f;
    return;
  }
  long base = ((long)(b * NTC + a)) * (SSC * SSC);
  __shared__ float red[8];
  float mx = -3.0e38f;
  for (int k = tid; k < SSC * SSC; k += 256) mx = fmaxf(mx, rule[base + k]);
  mx = wave_max(mx);
  if ((tid & 31) == 0) red[tid >> 5] = mx;
  __syncthreads();
  float bm = red[0];
#pragma unroll
  for (int q = 1; q < 8; ++q) bm = fmaxf(bm, red[q]);
  for (int k = tid; k < KP; k += 256) {
    int x = k / 96, c = k - x * 96;
    float e = (c < SSC) ? __expf(rule[base + x * SSC + c] - bm) : 0.0f;
    eruleH[hbase + k] = (_Float16)e;
  }
  if (tid == 0) mruleA[b * NTC + a] = bm;
}

// ---------------------------------------------------------------------------
// Fused width-w step. Block owns whole spans (rows span-major, stride P).
//   Phase 1: eL/eR = fp16 exp(span - max) in LDS, per-row scale = mL+mR.
//   Phase 2: 8 waves split 270 k-steps; per step x is wave-uniform
//            (k = x*96+c layout), A frag = splat(eL[x]) * contiguous eR run,
//            B frag = 32B aligned global load; 2 (or 4) WMMAs per step.
//   Phase 3: LDS cross-wave C reduction, then in-block scaled log-sum over
//            splits writes beta directly (no extra kernel).
// R = rows per block: 16 (w<=16) or 32 (w>16).
// ---------------------------------------------------------------------------
template <int R>
__global__ void __launch_bounds__(256)
k_cky_span(float* __restrict__ beta, const _Float16* __restrict__ eruleH,
           const float* __restrict__ mruleA, int w, int P, int SB) {
  const int ns   = NNC - w;
  const int tid  = threadIdx.x;
  const int lane = tid & 31;
  const int wv   = tid >> 5;
  const int b    = blockIdx.y;
  const int spanBase = blockIdx.x * SB;

  __shared__ __align__(16) _Float16 eL[R * ELSTR];
  __shared__ __align__(16) _Float16 eR[R * ERSTR];
  __shared__ float mLs[R], mRs[R], scaleL[R];
  __shared__ float dotL[R * 32];
  __shared__ float psum[(R / 16) * 2 * 8 * 256];

  // ---- phase 1 -------------------------------------------------------------
  for (int task = wv; task < 2 * R; task += 8) {
    int m = task >> 1, isR = task & 1;
    int sIdx = m / P, u = m - sIdx * P;
    int s = spanBase + sIdx;
    int valid = (s < ns) && (u < w);
    int ss = valid ? s : 0, uu = valid ? u : 0;
    const float* src = isR
        ? beta + (((long)(b * NNC) + (ss + uu + 1)) * NNC + (ss + w)) * SSC
        : beta + (((long)(b * NNC) + ss) * NNC + (ss + uu)) * SSC;
    float mx = -3.0e38f;
    for (int c = lane; c < SSC; c += 32) mx = fmaxf(mx, src[c]);
    mx = wave_max(mx);
    if (isR) {
      _Float16* dst = eR + m * ERSTR;
      for (int c = lane; c < 96; c += 32)
        dst[c] = (_Float16)((valid && c < SSC) ? __expf(src[c] - mx) : 0.0f);
      if (lane == 0) mRs[m] = mx;
    } else {
      _Float16* dst = eL + m * ELSTR;
      for (int c = lane; c < ELSTR; c += 32)
        dst[c] = (_Float16)((valid && c < SSC) ? __expf(src[c] - mx) : 0.0f);
      if (lane == 0) mLs[m] = mx;
    }
  }
  __syncthreads();
  if (tid < R) scaleL[tid] = mLs[tid] + mRs[tid];

  // ---- phase 2: WMMA k-loop ------------------------------------------------
  const int mlane = lane & 15;
  const int kbA = (lane < 16) ? 0 : 16;   // A 16-bit fragment K pattern (contiguous, matches B)
  const int kbB = (lane < 16) ? 0 : 16;   // B 16-bit fragment K pattern
  const long rowB0 = ((long)(b * 32) + mlane) * KP;   // a = 0..15
  const long rowB1 = rowB0 + (long)16 * KP;           // a = 16..31
  const _Float16* eL0 = eL + mlane * ELSTR;
  const _Float16* eR0 = eR + mlane * ERSTR;

  const v8f vzero = {0.f, 0.f, 0.f, 0.f, 0.f, 0.f, 0.f, 0.f};
  v8f acc00 = vzero, acc01 = vzero;
  v8f acc10 = vzero, acc11 = vzero;

  for (int ks = wv; ks < KSTEPS; ks += 8) {
    const int k0 = ks * 32;
    const int x  = ks / 3;               // wave-uniform symbol index
    const int c0 = (ks - x * 3) * 32;    // 0, 32, 64
    v16h bf0 = *(const v16h*)(eruleH + rowB0 + k0 + kbB);   // 32B aligned
    v16h bf1 = *(const v16h*)(eruleH + rowB1 + k0 + kbB);
    {
      _Float16 el = eL0[x];
      h8 e0 = *(const h8*)(eR0 + c0 + kbA);        // 16B aligned
      h8 e1 = *(const h8*)(eR0 + c0 + kbA + 8);
      h8 sp;
#pragma unroll
      for (int i = 0; i < 8; ++i) sp[i] = el;
      h8 p0 = e0 * sp, p1 = e1 * sp;
      v16h af;
#pragma unroll
      for (int j = 0; j < 8; ++j) { af[j] = p0[j]; af[j + 8] = p1[j]; }
      acc00 = __builtin_amdgcn_wmma_f32_16x16x32_f16(false, af, false, bf0,
                                                     (short)0, acc00, false, false);
      acc01 = __builtin_amdgcn_wmma_f32_16x16x32_f16(false, af, false, bf1,
                                                     (short)0, acc01, false, false);
    }
    if constexpr (R == 32) {             // second M-tile reuses B fragments
      _Float16 el = eL0[16 * ELSTR + x];
      h8 e0 = *(const h8*)(eR0 + 16 * ERSTR + c0 + kbA);
      h8 e1 = *(const h8*)(eR0 + 16 * ERSTR + c0 + kbA + 8);
      h8 sp;
#pragma unroll
      for (int i = 0; i < 8; ++i) sp[i] = el;
      h8 p0 = e0 * sp, p1 = e1 * sp;
      v16h af;
#pragma unroll
      for (int j = 0; j < 8; ++j) { af[j] = p0[j]; af[j + 8] = p1[j]; }
      acc10 = __builtin_amdgcn_wmma_f32_16x16x32_f16(false, af, false, bf0,
                                                     (short)0, acc10, false, false);
      acc11 = __builtin_amdgcn_wmma_f32_16x16x32_f16(false, af, false, bf1,
                                                     (short)0, acc11, false, false);
    }
  }

  // ---- phase 3: cross-wave reduction ---------------------------------------
#pragma unroll
  for (int v = 0; v < 8; ++v) {
    psum[((0 * 8 + wv) * 256) + lane * 8 + v] = acc00[v];
    psum[((1 * 8 + wv) * 256) + lane * 8 + v] = acc01[v];
    if constexpr (R == 32) {
      psum[((2 * 8 + wv) * 256) + lane * 8 + v] = acc10[v];
      psum[((3 * 8 + wv) * 256) + lane * 8 + v] = acc11[v];
    }
  }
  __syncthreads();
  {
    const int lane_ = tid >> 3, vv = tid & 7;
    const int nn = lane_ & 15;
    const int mm = vv + ((lane_ >= 16) ? 8 : 0);  // D layout: M = v (+8 upper)
#pragma unroll
    for (int mt = 0; mt < R / 16; ++mt) {
#pragma unroll
      for (int nt = 0; nt < 2; ++nt) {
        float sum = 0.0f;
#pragma unroll
        for (int q = 0; q < 8; ++q)
          sum += psum[(((mt * 2 + nt) * 8 + q) * 256) + lane_ * 8 + vv];
        dotL[(mt * 16 + mm) * 32 + (nt * 16 + nn)] = sum;
      }
    }
  }
  __syncthreads();

  // ---- fused combine: beta[b,s,s+w,a] --------------------------------------
  {
    const int a = tid & 31;
    for (int si = tid >> 5; si < SB; si += 8) {
      int s = spanBase + si;
      if (s >= ns) continue;
      float Cmax = -3.0e38f;
      for (int u = 0; u < w; ++u)
        Cmax = fmaxf(Cmax, scaleL[si * P + u]);
      if (a < NTC) {
        float sum = 0.0f;
        for (int u = 0; u < w; ++u)
          sum += __expf(scaleL[si * P + u] - Cmax) * dotL[(si * P + u) * 32 + a];
        float sc = __logf(fmaxf(sum, 1e-38f)) + Cmax + mruleA[b * NTC + a];
        beta[(((long)(b * NNC) + s) * NNC + (s + w)) * SSC + a] = sc;
      }
    }
  }
}

// ---------------------------------------------------------------------------
// log_Z[b] = logsumexp_a(beta[b,0,n-1,a] + root[b,a])
// ---------------------------------------------------------------------------
__global__ void k_finalize(const float* __restrict__ beta,
                           const float* __restrict__ root,
                           float* __restrict__ out) {
  int b = blockIdx.x;
  int a = threadIdx.x;
  float v = -3.0e38f;
  if (a < NTC)
    v = beta[(((long)(b * NNC) + 0) * NNC + (NNC - 1)) * SSC + a] + root[b * NTC + a];
  float mx = wave_max(v);
  float e  = (a < NTC) ? __expf(v - mx) : 0.0f;
  float sm = wave_sum(e);
  if (a == 0) out[b] = __logf(sm) + mx;
}

// ---------------------------------------------------------------------------
extern "C" void kernel_launch(void* const* d_in, const int* in_sizes, int n_in,
                              void* d_out, int out_size, void* d_ws, size_t ws_size,
                              hipStream_t stream) {
  (void)in_sizes; (void)n_in; (void)out_size; (void)ws_size;
  const float* unary = (const float*)d_in[0];
  const float* rule  = (const float*)d_in[1];
  const float* root  = (const float*)d_in[2];
  float* out = (float*)d_out;

  char* ws = (char*)d_ws;
  size_t off = 0;
  auto wsAlloc = [&](size_t bytes) -> void* {
    void* p = ws + off;
    off = (off + bytes + 255) & ~(size_t)255;
    return p;
  };
  float*    beta   = (float*)   wsAlloc((size_t)BBC * NNC * NNC * SSC * 4);  // 5.9 MB
  _Float16* eruleH = (_Float16*)wsAlloc((size_t)BBC * 32 * KP * 2);          // 8.9 MB
  float*    mruleA = (float*)   wsAlloc((size_t)BBC * NTC * 4);

  {
    int total = BBC * NNC * NNC * SSC;
    k_init_beta<<<(total + 255) / 256, 256, 0, stream>>>(unary, beta);
  }
  k_prep_rule<<<dim3(32, BBC), 256, 0, stream>>>(rule, eruleH, mruleA);

  for (int w = 1; w < NNC; ++w) {
    int ns = NNC - w;
    int P = 1;
    while (P < w) P <<= 1;          // next pow2 >= w
    if (w > 16) P = 32;
    if (w <= 16) {
      int SB = 16 / P;              // spans per block
      int G  = (ns + SB - 1) / SB;
      k_cky_span<16><<<dim3(G, BBC), 256, 0, stream>>>(beta, eruleH, mruleA, w, P, SB);
    } else {
      k_cky_span<32><<<dim3(ns, BBC), 256, 0, stream>>>(beta, eruleH, mruleA, w, P, 1);
    }
  }
  k_finalize<<<BBC, 32, 0, stream>>>(beta, root, out);
}
